// PuncCapBiLstmCrf_53214644798089
// MI455X (gfx1250) — matrix operand order repaired and checked by reading, not compiled, not run on hardware
//
#include <hip/hip_runtime.h>
#include <hip/hip_bf16.h>

typedef __bf16 bf16_t;
typedef __attribute__((ext_vector_type(16))) __bf16 v16bf;
typedef __attribute__((ext_vector_type(8)))  float  v8f;

#define Bn   16
#define Ln   256
#define Hn   768
#define HALF 384
#define G4   1536
#define NTOK 4096
#define NP   9
#define NC   3

__device__ __forceinline__ v8f zero8() {
  v8f v;
#pragma unroll
  for (int i = 0; i < 8; ++i) v[i] = 0.f;
  return v;
}

__device__ __forceinline__ v8f wmma_bf16(v16bf a, v16bf b, v8f c) {
  // D = A(16x32 bf16) * B(32x16 bf16) + C(16x16 f32)
  return __builtin_amdgcn_wmma_f32_16x16x32_bf16(false, a, false, b, (short)0, c,
                                                 false, false);
}

__device__ __forceinline__ float sigf(float x) { return 1.f / (1.f + __expf(-x)); }

// ---------------------------------------------------------------- converts
__global__ void k_f32_to_bf16(const float* __restrict__ src, bf16_t* __restrict__ dst, int n) {
  int i = blockIdx.x * 256 + threadIdx.x;
  if (i < n) dst[i] = (bf16_t)src[i];
}

// ---------------------------------------------------- embeddings + layernorm
__global__ void k_embed_ln(const int* __restrict__ ids, const int* __restrict__ tti,
                           const float* __restrict__ wemb, const float* __restrict__ pemb,
                           const float* __restrict__ temb, const float* __restrict__ g,
                           const float* __restrict__ bln, bf16_t* __restrict__ x0) {
  const int tok = blockIdx.x;           // b*L + l
  const int l   = tok & 255;
  const int tid = threadIdx.x;
  __shared__ float red[256];
  const int id = ids[tok];
  const int tt = tti[tok];
  float e[3];
#pragma unroll
  for (int j = 0; j < 3; ++j) {
    int d = tid + j * 256;
    e[j] = wemb[(long)id * Hn + d] + pemb[(long)l * Hn + d] + temb[(long)tt * Hn + d];
  }
  float s = e[0] + e[1] + e[2];
  red[tid] = s; __syncthreads();
  for (int o = 128; o > 0; o >>= 1) { if (tid < o) red[tid] += red[tid + o]; __syncthreads(); }
  const float mean = red[0] / 768.f;
  __syncthreads();
  float v = 0.f;
#pragma unroll
  for (int j = 0; j < 3; ++j) { float d = e[j] - mean; v += d * d; }
  red[tid] = v; __syncthreads();
  for (int o = 128; o > 0; o >>= 1) { if (tid < o) red[tid] += red[tid + o]; __syncthreads(); }
  const float inv = rsqrtf(red[0] / 768.f + 1e-12f);
#pragma unroll
  for (int j = 0; j < 3; ++j) {
    int d = tid + j * 256;
    x0[(long)tok * Hn + d] = (bf16_t)((e[j] - mean) * inv * g[d] + bln[d]);
  }
}

// --------------------------------------- gate pre-activation GEMM (WMMA bf16)
// gx[4096,1536] = X[4096,768]bf16 @ Wih^T + (bih + bhh)
// Block: 8 waves, each owns one M tile; all share a 64-column B panel that is
// double-buffered through LDS with gfx1250 async-to-LDS copies (ASYNCcnt).
// kc loop fully unrolled so accumulators stay pinned (no loop-carried copies).
__global__ void k_gemm_gx(const bf16_t* __restrict__ X, const bf16_t* __restrict__ W,
                          const float* __restrict__ bih, const float* __restrict__ bhh,
                          float* __restrict__ gx) {
  const int tid  = threadIdx.x;
  const int wave = tid >> 5;
  const int lane = tid & 31;
  const int lr   = lane & 15;
  const int half = lane >> 4;
  const int mt   = blockIdx.x * 8 + wave;   // M tile per wave (0..255)
  const int nb   = blockIdx.y * 64;         // shared 64-col N panel

  __shared__ __align__(32) bf16_t Bs[2][64 * 64];   // 16 KB double buffer

  // async staging pattern: thread -> row tid>>2 (0..63), 16 halves at (tid&3)*16
  const int srow  = tid >> 2;
  const int skoff = (tid & 3) * 16;
  const bf16_t* wsrc = W + (long)(nb + srow) * Hn + skoff;
  const unsigned lds0 = (unsigned)(uintptr_t)&Bs[0][srow * 64 + skoff];
  const unsigned lds1 = (unsigned)(uintptr_t)&Bs[1][srow * 64 + skoff];

  // prologue: chunk 0 -> buffer 0
  asm volatile("global_load_async_to_lds_b128 %0, %1, off"
               :: "v"(lds0), "v"(wsrc) : "memory");
  asm volatile("global_load_async_to_lds_b128 %0, %1, off"
               :: "v"(lds0 + 16u), "v"(wsrc + 8) : "memory");

  v8f acc[4];
#pragma unroll
  for (int ti = 0; ti < 4; ++ti) acc[ti] = zero8();
  const bf16_t* ap = X + (long)(mt * 16 + lr) * Hn + half * 16;

#pragma unroll
  for (int kc = 0; kc < 12; ++kc) {               // 12 chunks of K=64
    const int cur = kc & 1;
    if (kc + 1 < 12) {                            // issue next chunk first
      const bf16_t* gsrc = wsrc + (kc + 1) * 64;
      const unsigned ldsn = (cur == 0) ? lds1 : lds0;
      asm volatile("global_load_async_to_lds_b128 %0, %1, off"
                   :: "v"(ldsn), "v"(gsrc) : "memory");
      asm volatile("global_load_async_to_lds_b128 %0, %1, off"
                   :: "v"(ldsn + 16u), "v"(gsrc + 8) : "memory");
      asm volatile("s_wait_asynccnt 2" ::: "memory");   // current chunk landed
    } else {
      asm volatile("s_wait_asynccnt 0" ::: "memory");
    }
    __syncthreads();                              // panel visible to all waves
    __builtin_prefetch(ap + (kc + 1) * 64, 0, 1); // global_prefetch_b8 for A
#pragma unroll
    for (int kk = 0; kk < 64; kk += 32) {
      v16bf a = *(const v16bf*)(ap + kc * 64 + kk);
#pragma unroll
      for (int ti = 0; ti < 4; ++ti) {
        v16bf bf = *(const v16bf*)&Bs[cur][(ti * 16 + lr) * 64 + kk + half * 16];
        acc[ti] = wmma_bf16(a, bf, acc[ti]);
      }
    }
    __syncthreads();                              // done with buf before reuse
  }
#pragma unroll
  for (int ti = 0; ti < 4; ++ti)
#pragma unroll
    for (int r = 0; r < 8; ++r) {
      const int row = mt * 16 + r + 8 * half;
      const int col = nb + ti * 16 + lr;
      gx[(long)row * G4 + col] = acc[ti][r] + bih[col] + bhh[col];
    }
}

// ------------------------------------------ persistent BiLSTM recurrence
// grid.x = 2 (fwd, bwd). 8 waves; each wave owns 48 hidden units = 12 gate
// tiles (3 per gate). h lives in LDS (bf16), c lives in registers.
__global__ void k_lstm(const float* __restrict__ gxF, const float* __restrict__ gxB,
                       const bf16_t* __restrict__ whhF, const bf16_t* __restrict__ whhB,
                       bf16_t* __restrict__ xout) {
  const bool fwd      = (blockIdx.x == 0);
  const float* gx     = fwd ? gxF : gxB;
  const bf16_t* whh   = fwd ? whhF : whhB;
  const int coloff    = fwd ? 0 : HALF;

  const int tid  = threadIdx.x;
  const int wave = tid >> 5;
  const int lane = tid & 31;
  const int lr   = lane & 15;
  const int half = lane >> 4;
  const int u0   = wave * 48;

  __shared__ __align__(32) bf16_t hbuf[Bn * HALF];   // 12 KB, h_t as bf16
  for (int i = tid; i < Bn * HALF; i += 256) hbuf[i] = (bf16_t)0.f;
  __syncthreads();

  float cst[3][8];
#pragma unroll
  for (int ti = 0; ti < 3; ++ti)
#pragma unroll
    for (int r = 0; r < 8; ++r) cst[ti][r] = 0.f;

  for (int t = 0; t < Ln; ++t) {
    const int tt = fwd ? t : (Ln - 1 - t);
    v8f acc[4][3];
#pragma unroll
    for (int g = 0; g < 4; ++g)
#pragma unroll
      for (int ti = 0; ti < 3; ++ti) acc[g][ti] = zero8();

#pragma unroll
    for (int kb = 0; kb < HALF; kb += 32) {
      v16bf a = *(const v16bf*)&hbuf[lr * HALF + kb + half * 16];   // ds_load
#pragma unroll
      for (int g = 0; g < 4; ++g)
#pragma unroll
        for (int ti = 0; ti < 3; ++ti) {
          const int rw = g * HALF + u0 + ti * 16 + lr;              // Whh row
          v16bf bf = *(const v16bf*)&whh[(long)rw * HALF + kb + half * 16];
          acc[g][ti] = wmma_bf16(a, bf, acc[g][ti]);
        }
    }
    __syncthreads();   // all waves done reading hbuf for this step
#pragma unroll
    for (int ti = 0; ti < 3; ++ti)
#pragma unroll
      for (int r = 0; r < 8; ++r) {
        const int m = r + 8 * half;                 // batch row
        const int j = u0 + ti * 16 + lr;            // hidden unit
        const long base = (long)(m * Ln + tt) * G4;
        const float ip = acc[0][ti][r] + gx[base + 0 * HALF + j];
        const float fp = acc[1][ti][r] + gx[base + 1 * HALF + j];
        const float gp = acc[2][ti][r] + gx[base + 2 * HALF + j];
        const float op = acc[3][ti][r] + gx[base + 3 * HALF + j];
        const float ig = sigf(ip), fg = sigf(fp), og = sigf(op);
        const float gg = tanhf(gp);
        const float c  = fg * cst[ti][r] + ig * gg;
        cst[ti][r] = c;
        const float h = og * tanhf(c);
        hbuf[m * HALF + j] = (bf16_t)h;
        xout[(long)(m * Ln + tt) * Hn + coloff + j] = (bf16_t)h;
      }
    __syncthreads();   // hbuf fully updated before next step
  }
}

// ---------------------------------------------- valid_ids prefix compaction
__global__ void k_prefix(const int* __restrict__ valid, int* __restrict__ tgt) {
  const int b = blockIdx.x, tid = threadIdx.x;
  __shared__ int s[256];
  const int v = valid[b * Ln + tid];
  s[tid] = v; __syncthreads();
  for (int o = 1; o < 256; o <<= 1) {
    int x = (tid >= o) ? s[tid - o] : 0;
    __syncthreads();
    s[tid] += x;
    __syncthreads();
  }
  tgt[b * Ln + tid] = v ? (s[tid] - 1) : Ln;   // Ln == dropped
}

__global__ void k_init_logits(float* __restrict__ logits, const float* __restrict__ bias,
                              int T, int total) {
  int i = blockIdx.x * 256 + threadIdx.x;
  if (i < total) logits[i] = bias[i % T];
}

__global__ void k_zero(float* __restrict__ out) {
  if (blockIdx.x == 0 && threadIdx.x == 0) out[0] = 0.f;
}

// ------------------------------------------------- heads at valid positions
__global__ void k_heads(const bf16_t* __restrict__ seq, const int* __restrict__ valid,
                        const int* __restrict__ tgt, const float* __restrict__ pW,
                        const float* __restrict__ pb, const float* __restrict__ cW,
                        const float* __restrict__ cb, float* __restrict__ plog,
                        float* __restrict__ clog) {
  const int tok = blockIdx.x;
  if (!valid[tok]) return;                 // uniform per block
  const int tg = tgt[tok];
  if (tg >= Ln) return;
  const int b = tok >> 8;
  const int tid = threadIdx.x;
  __shared__ float red[256];
  const float x0 = (float)seq[(long)tok * Hn + tid];
  const float x1 = (float)seq[(long)tok * Hn + tid + 256];
  const float x2 = (float)seq[(long)tok * Hn + tid + 512];
  for (int row = 0; row < NP + NC; ++row) {
    const float* w = (row < NP) ? (pW + (long)row * Hn) : (cW + (long)(row - NP) * Hn);
    red[tid] = x0 * w[tid] + x1 * w[tid + 256] + x2 * w[tid + 512];
    __syncthreads();
    for (int o = 128; o > 0; o >>= 1) { if (tid < o) red[tid] += red[tid + o]; __syncthreads(); }
    if (tid == 0) {
      if (row < NP) plog[((long)b * Ln + tg) * NP + row]        = red[0] + pb[row];
      else          clog[((long)b * Ln + tg) * NC + (row - NP)] = red[0] + cb[row - NP];
    }
    __syncthreads();
  }
}

// ---------------------------------------- CRF: Viterbi + forward LLH + loss
__global__ void k_crf(const float* __restrict__ pem, const float* __restrict__ cem,
                      const float* __restrict__ pst, const float* __restrict__ pen,
                      const float* __restrict__ ptr_, const float* __restrict__ cst_,
                      const float* __restrict__ cen, const float* __restrict__ ctr,
                      const int* __restrict__ plab, const int* __restrict__ clab,
                      const int* __restrict__ mask, int* __restrict__ phist,
                      int* __restrict__ chist, float* __restrict__ out) {
  const bool isP = (blockIdx.x == 0);
  const int T = isP ? NP : NC;
  const float* em = isP ? pem : cem;
  const float* st = isP ? pst : cst_;
  const float* en = isP ? pen : cen;
  const float* tr = isP ? ptr_ : ctr;
  const int* lab  = isP ? plab : clab;
  int* hist       = isP ? phist : chist;
  float* tago     = out + 1 + (isP ? 0 : NTOK);

  const int tid = threadIdx.x;
  __shared__ float vs[2][Bn][NP];
  __shared__ float al[2][Bn][NP];
  __shared__ float lred[Bn];

  const bool act = tid < Bn * T;
  const int b   = act ? tid / T : 0;
  const int cur = act ? tid % T : 0;
  if (act) {
    const float e0 = em[((long)b * Ln) * T + cur];
    vs[0][b][cur] = st[cur] + e0;
    al[0][b][cur] = st[cur] + e0;
  }
  __syncthreads();
  for (int t = 1; t < Ln; ++t) {
    const int pp = (t - 1) & 1, np = t & 1;
    if (act) {
      const float et = em[((long)b * Ln + t) * T + cur];
      const int m = mask[b * Ln + t];
      float best = -3.0e38f; int bi = 0;
      float mx = -3.0e38f;
      for (int p = 0; p < T; ++p) {
        const float tv = tr[p * T + cur];
        const float v = vs[pp][b][p] + tv;
        if (v > best) { best = v; bi = p; }
        const float a = al[pp][b][p] + tv;
        if (a > mx) mx = a;
      }
      float s = 0.f;
      for (int p = 0; p < T; ++p) s += __expf(al[pp][b][p] + tr[p * T + cur] - mx);
      hist[((long)(t - 1) * Bn + b) * T + cur] = bi;
      vs[np][b][cur] = m ? (best + et) : vs[pp][b][cur];
      al[np][b][cur] = m ? (mx + __logf(s) + et) : al[pp][b][cur];
    }
    __syncthreads();
  }
  __threadfence_block();
  __syncthreads();
  if (tid < Bn) {
    const int bb = tid;
    float mx = -3.0e38f;
    for (int p = 0; p < T; ++p) { float a = al[1][bb][p] + en[p]; if (a > mx) mx = a; }
    float s = 0.f;
    for (int p = 0; p < T; ++p) s += __expf(al[1][bb][p] + en[p] - mx);
    const float logZ = mx + __logf(s);

    float best = -3.0e38f; int last = 0;
    for (int p = 0; p < T; ++p) {
      float v = vs[1][bb][p] + en[p];
      if (v > best) { best = v; last = p; }
    }
    // numerator
    const int t0 = lab[bb * Ln];
    float score = st[t0] + em[((long)bb * Ln) * T + t0];
    int nv = mask[bb * Ln];
    for (int t = 1; t < Ln; ++t) {
      const int m  = mask[bb * Ln + t];
      const int tp = lab[bb * Ln + t - 1], tc = lab[bb * Ln + t];
      if (m) score += tr[tp * T + tc] + em[((long)bb * Ln + t) * T + tc];
      nv += m;
    }
    score += en[lab[bb * Ln + nv - 1]];
    lred[bb] = score - logZ;
    // backtrack
    int tag = last;
    tago[bb * Ln + (Ln - 1)] = (float)(tag * mask[bb * Ln + (Ln - 1)]);
    for (int t = Ln - 2; t >= 0; --t) {
      if (mask[bb * Ln + t + 1]) tag = hist[((long)t * Bn + bb) * T + tag];
      tago[bb * Ln + t] = (float)(tag * mask[bb * Ln + t]);
    }
  }
  __syncthreads();
  if (tid == 0) {
    float s = 0.f;
    for (int i = 0; i < Bn; ++i) s += lred[i];
    atomicAdd(out, -s);
  }
}

// --------------------------------------------------------------- launcher
extern "C" void kernel_launch(void* const* d_in, const int* in_sizes, int n_in,
                              void* d_out, int out_size, void* d_ws, size_t ws_size,
                              hipStream_t stream) {
  (void)in_sizes; (void)n_in; (void)out_size; (void)ws_size;
  const int* input_ids  = (const int*)d_in[0];
  const int* token_type = (const int*)d_in[1];
  const int* valid      = (const int*)d_in[2];
  const int* lmask      = (const int*)d_in[3];
  const int* plab       = (const int*)d_in[4];
  const int* clab       = (const int*)d_in[5];
  const float* wemb = (const float*)d_in[6];
  const float* pemb = (const float*)d_in[7];
  const float* temb = (const float*)d_in[8];
  const float* lng  = (const float*)d_in[9];
  const float* lnb  = (const float*)d_in[10];
  const float *Wih[4], *Whh[4], *bih[4], *bhh[4];
  for (int i = 0; i < 4; ++i) {                 // (L1 fwd, L1 bwd, L2 fwd, L2 bwd)
    Wih[i] = (const float*)d_in[11 + i * 4 + 0];
    Whh[i] = (const float*)d_in[11 + i * 4 + 1];
    bih[i] = (const float*)d_in[11 + i * 4 + 2];
    bhh[i] = (const float*)d_in[11 + i * 4 + 3];
  }
  const float* pW = (const float*)d_in[27];
  const float* pb = (const float*)d_in[28];
  const float* cW = (const float*)d_in[29];
  const float* cb = (const float*)d_in[30];
  const float* pstart = (const float*)d_in[31];
  const float* pend   = (const float*)d_in[32];
  const float* ptrans = (const float*)d_in[33];
  const float* cstart = (const float*)d_in[34];
  const float* cend   = (const float*)d_in[35];
  const float* ctrans = (const float*)d_in[36];

  char* w = (char*)d_ws;
  auto alloc = [&](size_t bytes) { char* p = w; w += (bytes + 255) & ~(size_t)255; return p; };
  bf16_t* x0  = (bf16_t*)alloc((size_t)NTOK * Hn * 2);
  bf16_t* x1  = (bf16_t*)alloc((size_t)NTOK * Hn * 2);
  bf16_t* x2  = (bf16_t*)alloc((size_t)NTOK * Hn * 2);
  float*  gx0 = (float*)alloc((size_t)NTOK * G4 * 4);
  float*  gx1 = (float*)alloc((size_t)NTOK * G4 * 4);
  bf16_t* wihb[4]; bf16_t* whhb[4];
  for (int i = 0; i < 4; ++i) {
    wihb[i] = (bf16_t*)alloc((size_t)G4 * Hn * 2);
    whhb[i] = (bf16_t*)alloc((size_t)G4 * HALF * 2);
  }
  float* plog  = (float*)alloc((size_t)NTOK * NP * 4);
  float* clog  = (float*)alloc((size_t)NTOK * NC * 4);
  int*   tgt   = (int*)alloc((size_t)NTOK * 4);
  int*   phist = (int*)alloc((size_t)(Ln - 1) * Bn * NP * 4);
  int*   chist = (int*)alloc((size_t)(Ln - 1) * Bn * NC * 4);

  for (int i = 0; i < 4; ++i) {
    k_f32_to_bf16<<<dim3((G4 * Hn + 255) / 256), 256, 0, stream>>>(Wih[i], wihb[i], G4 * Hn);
    k_f32_to_bf16<<<dim3((G4 * HALF + 255) / 256), 256, 0, stream>>>(Whh[i], whhb[i], G4 * HALF);
  }
  k_embed_ln<<<dim3(NTOK), 256, 0, stream>>>(input_ids, token_type, wemb, pemb, temb, lng, lnb, x0);

  // layer 1
  k_gemm_gx<<<dim3(NTOK / 128, G4 / 64), 256, 0, stream>>>(x0, wihb[0], bih[0], bhh[0], gx0);
  k_gemm_gx<<<dim3(NTOK / 128, G4 / 64), 256, 0, stream>>>(x0, wihb[1], bih[1], bhh[1], gx1);
  k_lstm<<<dim3(2), 256, 0, stream>>>(gx0, gx1, whhb[0], whhb[1], x1);
  // layer 2
  k_gemm_gx<<<dim3(NTOK / 128, G4 / 64), 256, 0, stream>>>(x1, wihb[2], bih[2], bhh[2], gx0);
  k_gemm_gx<<<dim3(NTOK / 128, G4 / 64), 256, 0, stream>>>(x1, wihb[3], bih[3], bhh[3], gx1);
  k_lstm<<<dim3(2), 256, 0, stream>>>(gx0, gx1, whhb[2], whhb[3], x2);

  k_prefix<<<dim3(Bn), 256, 0, stream>>>(valid, tgt);
  k_init_logits<<<dim3((NTOK * NP + 255) / 256), 256, 0, stream>>>(plog, pb, NP, NTOK * NP);
  k_init_logits<<<dim3((NTOK * NC + 255) / 256), 256, 0, stream>>>(clog, cb, NC, NTOK * NC);
  k_zero<<<dim3(1), 1, 0, stream>>>((float*)d_out);
  k_heads<<<dim3(NTOK), 256, 0, stream>>>(x2, valid, tgt, pW, pb, cW, cb, plog, clog);
  k_crf<<<dim3(2), 256, 0, stream>>>(plog, clog, pstart, pend, ptrans, cstart, cend, ctrans,
                                     plab, clab, lmask, phist, chist, (float*)d_out);
}
